// AdjacencyMatrixLearning_40097814675713
// MI455X (gfx1250) — compile-verified, matrix operand order checked by reading, not run on hardware
//
#include <hip/hip_runtime.h>
#include <hip/hip_bf16.h>

// ---------------------------------------------------------------------------
// AdjacencyMatrixLearning on gfx1250 (MI455X), bf16 WMMA path.
// B=4096, C=22, T=9, HID=64.  One wave32 per (b,t) problem.
// Prepass converts W1..W4 (f32) -> bf16 in d_ws once; main kernel loads
// weight fragments as raw b128 dwords.  Bias folded into WMMA accumulator
// init; LDS addressing uses hoisted lane bases (ds immediate offsets).
// Padded rows (22..31) are address-clamped only -- their garbage is
// quarantined in A rows/cols >= 22, which softmax/output never read.
// ---------------------------------------------------------------------------

typedef __attribute__((ext_vector_type(16))) __bf16 v16bf;
typedef __attribute__((ext_vector_type(16))) float  v16f;
typedef __attribute__((ext_vector_type(8)))  float  v8f;

#define K_B 4096
#define K_C 22
#define K_T 9
#define K_HID 64

// ws layout (bf16 elements): W1[64*64] | W2[64*64] | W3[64*128] | W4[64*128]
#define WS_W1 0
#define WS_W2 4096
#define WS_W3 8192
#define WS_W4 16384
#define WS_TOTAL 24576

#define STG_STRIDE 66          // ushort stride of staging rows (bank-friendly)
#define STG_TILE   (16 * STG_STRIDE)

union FragU { v16bf bf; unsigned int ui[8]; };

// ---------------------------------------------------------------------------
// Prepass: f32 weights -> bf16 workspace (one conversion total, not per wave)
// ---------------------------------------------------------------------------
__global__ __launch_bounds__(256)
void convert_weights_kernel(const float* __restrict__ W1, const float* __restrict__ W2,
                            const float* __restrict__ W3, const float* __restrict__ W4,
                            __bf16* __restrict__ ws)
{
    const int i = (int)blockIdx.x * 256 + (int)threadIdx.x; // 0 .. 24575
    float v;
    if (i < WS_W2)       v = W1[i];
    else if (i < WS_W3)  v = W2[i - WS_W2];
    else if (i < WS_W4)  v = W3[i - WS_W3];
    else                 v = W4[i - WS_W4];
    ws[i] = (__bf16)v;
}

// ---------------------------------------------------------------------------
// Fragment helpers.  16-bit A/B fragment K-pattern (wave32):
//   lanes 0-15 hold K = {0..7,16..23}+kb, lanes 16-31 K += 8
//   packed element 2j -> kb + (j<4 ? 2j : 2j+8); element 2j+1 -> +1
// ---------------------------------------------------------------------------

// A-fragment of X: logical row-major [C x Kd], memory layout [c][d][t]
// (d-stride = T floats).  Rows >= 22 are clamped to row 0 (in-bounds dup;
// resulting garbage lives only in A rows/cols >= 22, never read).
__device__ __forceinline__ v16bf x_frag(const float* __restrict__ base,
                                        int t, int strideD, int mtile, int kbase) {
    const unsigned l = threadIdx.x & 31u;
    const int row  = mtile * 16 + (int)(l & 15u);
    const int rowc = row < K_C ? row : 0;
    const int kb   = kbase + (int)((l >> 4) * 8u);
    const float* p = base + (size_t)(rowc * strideD) * K_T + t;
    v16f xv;
#pragma unroll
    for (int j = 0; j < 8; ++j) {
        const int k0 = kb + (j < 4 ? 2 * j : 2 * j + 8);
        xv[2 * j]     = p[(size_t)k0 * K_T];
        xv[2 * j + 1] = p[(size_t)(k0 + 1) * K_T];
    }
    return __builtin_convertvector(xv, v16bf); // 8x v_cvt_pk_bf16_f32
}

// B-fragment of W^T from pre-converted bf16 weights (row-major [HID x Kd]).
// Dword indices {0..3} and {8..11} (+kb) are contiguous -> two b128 loads.
__device__ __forceinline__ v16bf wbf_frag(const __bf16* __restrict__ Wb,
                                          int strideD, int ntile, int kbase) {
    const unsigned l = threadIdx.x & 31u;
    const int col = ntile * 16 + (int)(l & 15u);
    const int kb  = kbase + (int)((l >> 4) * 8u);
    const unsigned int* p = (const unsigned int*)(Wb + (size_t)col * strideD + kb);
    FragU f;
#pragma unroll
    for (int j = 0; j < 8; ++j)
        f.ui[j] = p[j < 4 ? j : j + 4];
    return f.bf;
}

// A- or B-fragment from bf16 LDS staging.  laneRowOff = (l&15)*STG_STRIDE,
// hb4 = (l>>4)*4.  tile/kbase are compile-time -> ds immediate offsets.
__device__ __forceinline__ v16bf s_frag(const __bf16* S, int tile, int kbase,
                                        int laneRowOff, int hb4) {
    const unsigned int* p =
        (const unsigned int*)(S + tile * STG_TILE + kbase + laneRowOff) + hb4;
    FragU f;
#pragma unroll
    for (int j = 0; j < 8; ++j)
        f.ui[j] = p[j < 4 ? j : j + 4];
    return f.bf;
}

// ReLU + cvt, scatter D-tile into staging.  laneOff = (l>>4)*8*STG_STRIDE+(l&15);
// mt/nt give a compile-time tile origin, r*STG_STRIDE are immediate offsets.
__device__ __forceinline__ void store_act(__bf16* S, int mtile, int ntile,
                                          int laneOff, v8f acc) {
    __bf16* p = S + (mtile * STG_TILE + ntile * 16) + laneOff;
#pragma unroll
    for (int r = 0; r < 8; ++r)
        p[r * STG_STRIDE] = (__bf16)fmaxf(acc[r], 0.0f);
}

// Bias-splat accumulator: C[row][col] = bias[col], col fixed per lane.
__device__ __forceinline__ v8f bias_acc(float bia) {
    v8f a;
#pragma unroll
    for (int r = 0; r < 8; ++r) a[r] = bia;
    return a;
}

#define WMMA_BF16(A, Bm, Cm) \
    __builtin_amdgcn_wmma_f32_16x16x32_bf16(false, (A), false, (Bm), (short)0, (Cm), false, false)

__global__ __launch_bounds__(32)
void adj_softmax_kernel(const float* __restrict__ freq,   // [B,22,64,9]
                        const float* __restrict__ timeo,  // [B,22,128,9]
                        const __bf16* __restrict__ wsW,   // bf16 W1|W2|W3|W4
                        const float* __restrict__ b1, const float* __restrict__ b2,
                        const float* __restrict__ b3, const float* __restrict__ b4,
                        float* __restrict__ out)          // [B,9,22,22]
{
    __shared__ __bf16 stgF1[32 * STG_STRIDE];
    __shared__ __bf16 stgF2[32 * STG_STRIDE];
    __shared__ __bf16 stgT1[32 * STG_STRIDE];
    __shared__ __bf16 stgT2[32 * STG_STRIDE];
    __shared__ float  stgA[32 * 32];

    const int bt = (int)blockIdx.x;
    const int b  = bt / K_T;
    const int t  = bt - b * K_T;
    const unsigned l = threadIdx.x & 31u;
    const int lc  = (int)(l & 15u);
    const int hb  = (int)(l >> 4);

    // hoisted per-lane LDS bases
    const int stLaneOff  = hb * 8 * STG_STRIDE + lc;  // store side
    const int ldLaneRow  = lc * STG_STRIDE;           // load side (row gather)
    const int hb4        = hb * 4;                    // load side dword half
    const int aLaneOff   = hb * 8 * 32 + lc;          // stgA store side

    const float* Xf = freq  + (size_t)b * (K_C * 64  * K_T);
    const float* Xt = timeo + (size_t)b * (K_C * 128 * K_T);
    const __bf16* Wb1 = wsW + WS_W1;
    const __bf16* Wb2 = wsW + WS_W2;
    const __bf16* Wb3 = wsW + WS_W3;
    const __bf16* Wb4 = wsW + WS_W4;

    const v8f vzero = {0.f, 0.f, 0.f, 0.f, 0.f, 0.f, 0.f, 0.f};

    // ---- Phase F: F1 = relu(Xf W1^T + b1), F2 = relu(Xf W2^T + b2) ----
#pragma unroll
    for (int nt = 0; nt < 4; ++nt) {
        const v16bf w1a = wbf_frag(Wb1, 64, nt, 0),  w1b = wbf_frag(Wb1, 64, nt, 32);
        const v16bf w2a = wbf_frag(Wb2, 64, nt, 0),  w2b = wbf_frag(Wb2, 64, nt, 32);
        const float bia1 = b1[nt * 16 + lc];
        const float bia2 = b2[nt * 16 + lc];
#pragma unroll
        for (int mt = 0; mt < 2; ++mt) {
            const v16bf xa = x_frag(Xf, t, 64, mt, 0);
            const v16bf xb = x_frag(Xf, t, 64, mt, 32);
            v8f acc1 = bias_acc(bia1);
            v8f acc2 = bias_acc(bia2);
            acc1 = WMMA_BF16(xa, w1a, acc1);
            acc1 = WMMA_BF16(xb, w1b, acc1);
            acc2 = WMMA_BF16(xa, w2a, acc2);
            acc2 = WMMA_BF16(xb, w2b, acc2);
            store_act(stgF1, mt, nt, stLaneOff, acc1);
            store_act(stgF2, mt, nt, stLaneOff, acc2);
        }
    }

    // ---- Phase T: T1 = relu(Xt W3^T + b3), T2 = relu(Xt W4^T + b4), K=128 ----
#pragma unroll
    for (int nt = 0; nt < 4; ++nt) {
        v16bf w3f[4], w4f[4];
#pragma unroll
        for (int ks = 0; ks < 4; ++ks) {
            w3f[ks] = wbf_frag(Wb3, 128, nt, ks * 32);
            w4f[ks] = wbf_frag(Wb4, 128, nt, ks * 32);
        }
        const float bia3 = b3[nt * 16 + lc];
        const float bia4 = b4[nt * 16 + lc];
#pragma unroll
        for (int mt = 0; mt < 2; ++mt) {
            v8f a3 = bias_acc(bia3);
            v8f a4 = bias_acc(bia4);
#pragma unroll
            for (int ks = 0; ks < 4; ++ks) {
                const v16bf xk = x_frag(Xt, t, 128, mt, ks * 32);
                a3 = WMMA_BF16(xk, w3f[ks], a3);
                a4 = WMMA_BF16(xk, w4f[ks], a4);
            }
            store_act(stgT1, mt, nt, stLaneOff, a3);
            store_act(stgT2, mt, nt, stLaneOff, a4);
        }
    }

    // Same-wave LDS ops are in-order; just stop the compiler reordering.
    __asm__ __volatile__("" ::: "memory");

    // ---- Phase A: A = (F1 F2^T + T1 T2^T) / sqrt(22) ----
    const float sc = 0.21320071635561041f; // 1/sqrt(22)
#pragma unroll
    for (int mt = 0; mt < 2; ++mt) {
        const v16bf f1a = s_frag(stgF1, mt, 0,  ldLaneRow, hb4);
        const v16bf f1b = s_frag(stgF1, mt, 32, ldLaneRow, hb4);
        const v16bf t1a = s_frag(stgT1, mt, 0,  ldLaneRow, hb4);
        const v16bf t1b = s_frag(stgT1, mt, 32, ldLaneRow, hb4);
#pragma unroll
        for (int nt = 0; nt < 2; ++nt) {
            const v16bf f2a = s_frag(stgF2, nt, 0,  ldLaneRow, hb4);
            const v16bf f2b = s_frag(stgF2, nt, 32, ldLaneRow, hb4);
            const v16bf t2a = s_frag(stgT2, nt, 0,  ldLaneRow, hb4);
            const v16bf t2b = s_frag(stgT2, nt, 32, ldLaneRow, hb4);
            v8f acc = vzero;
            acc = WMMA_BF16(f1a, f2a, acc);
            acc = WMMA_BF16(f1b, f2b, acc);
            acc = WMMA_BF16(t1a, t2a, acc);
            acc = WMMA_BF16(t1b, t2b, acc);
            float* pA = stgA + (mt * 16 * 32 + nt * 16) + aLaneOff;
#pragma unroll
            for (int r = 0; r < 8; ++r)
                pA[r * 32] = acc[r] * sc;
        }
    }
    __asm__ __volatile__("" ::: "memory");

    // ---- Softmax over e (last axis), lanes 0..21 each own one row c ----
    if (l < (unsigned)K_C) {
        float v[K_C];
        float m = -1e30f;
#pragma unroll
        for (int e = 0; e < K_C; ++e) {
            v[e] = stgA[(int)l * 32 + e];
            m = fmaxf(m, v[e]);
        }
        float s = 0.0f;
#pragma unroll
        for (int e = 0; e < K_C; ++e) {
            v[e] = __expf(v[e] - m);
            s += v[e];
        }
        const float inv = 1.0f / s;
        float* o = out + ((size_t)bt * K_C + l) * K_C;
#pragma unroll
        for (int e = 0; e < K_C; ++e)
            o[e] = v[e] * inv;
    }
}

extern "C" void kernel_launch(void* const* d_in, const int* in_sizes, int n_in,
                              void* d_out, int out_size, void* d_ws, size_t ws_size,
                              hipStream_t stream) {
    const float* freq  = (const float*)d_in[0];
    const float* timeo = (const float*)d_in[1];
    const float* W1 = (const float*)d_in[2];
    const float* b1 = (const float*)d_in[3];
    const float* W2 = (const float*)d_in[4];
    const float* b2 = (const float*)d_in[5];
    const float* W3 = (const float*)d_in[6];
    const float* b3 = (const float*)d_in[7];
    const float* W4 = (const float*)d_in[8];
    const float* b4 = (const float*)d_in[9];
    float* out = (float*)d_out;
    __bf16* wsW = (__bf16*)d_ws;   // needs WS_TOTAL*2 = 48 KB of scratch

    // Prepass: weights f32 -> bf16 (96 * 256 = 24576 threads, one each)
    convert_weights_kernel<<<dim3(WS_TOTAL / 256), dim3(256), 0, stream>>>(
        W1, W2, W3, W4, wsW);

    // Main: one wave32 per (b,t) problem
    adj_softmax_kernel<<<dim3(K_B * K_T), dim3(32), 0, stream>>>(
        freq, timeo, wsW, b1, b2, b3, b4, out);
}
